// CapsuleNet_50036368998940
// MI455X (gfx1250) — matrix operand order repaired and checked by reading, not compiled, not run on hardware
//
#include <hip/hip_runtime.h>

// ---------------------------------------------------------------------------
// CapsuleNet forward for MI455X (gfx1250, wave32, WMMA + TDM).
// Heavy GEMMs (PrimaryCaps implicit-GEMM conv, routing prediction einsum) use
// v_wmma_f32_16x16x32_bf16; the contiguous weight tile is DMA'd into LDS by
// the Tensor Data Mover (tensor_load_to_lds + s_wait_tensorcnt) when the
// toolchain exposes it. All big streamed tensors are bf16 to respect the
// 23.3 TB/s HBM roofline; accumulation is fp32.
// ---------------------------------------------------------------------------

typedef __attribute__((ext_vector_type(16))) __bf16 v16bf;
typedef __attribute__((ext_vector_type(8)))  float  v8f;

union Frag16 {
  v16bf v;
  unsigned short s[16];
  uint4 q[2];
};

#if defined(__HIP_DEVICE_COMPILE__) && __has_builtin(__builtin_amdgcn_tensor_load_to_lds)
#define USE_TDM 1
typedef __attribute__((ext_vector_type(4))) unsigned int u32x4;
typedef __attribute__((ext_vector_type(4))) int          i32x4;
typedef __attribute__((ext_vector_type(8))) int          i32x8;
#else
#define USE_TDM 0
#endif

__device__ __forceinline__ unsigned short f2bf(float f) {
  unsigned int u = __float_as_uint(f);
  u += 0x7FFFu + ((u >> 16) & 1u);            // round-to-nearest-even
  return (unsigned short)(u >> 16);
}
__device__ __forceinline__ float bf2f(unsigned short b) {
  return __uint_as_float(((unsigned int)b) << 16);
}

// ---------------------------------------------------------------------------
// elementwise: fp32 -> bf16
__global__ void k_f32_to_bf16(const float* __restrict__ src,
                              unsigned short* __restrict__ dst, int n) {
  int i = blockIdx.x * 256 + threadIdx.x;
  if (i < n) dst[i] = f2bf(src[i]);
}

__global__ void k_zero(float* __restrict__ p, int n) {
  int i = blockIdx.x * 256 + threadIdx.x;
  if (i < n) p[i] = 0.0f;
}

// ---------------------------------------------------------------------------
// conv1: [32,1,64,64] -> relu -> h bf16 [32,64,56,56]   (~1 GF, VALU)
__global__ void k_conv1_relu(const float* __restrict__ x,
                             const float* __restrict__ w,
                             const float* __restrict__ bias,
                             unsigned short* __restrict__ h) {
  int idx = blockIdx.x * 256 + threadIdx.x;
  if (idx >= 32 * 64 * 56 * 56) return;
  int ox = idx % 56;
  int oy = (idx / 56) % 56;
  int oc = (idx / 3136) % 64;
  int b  = idx / 200704;
  const float* xin = x + b * 4096;
  const float* wk  = w + oc * 81;
  float acc = bias[oc];
  #pragma unroll
  for (int ky = 0; ky < 9; ++ky) {
    const float* xr = xin + (oy + ky) * 64 + ox;
    #pragma unroll
    for (int kx = 0; kx < 9; ++kx) acc += xr[kx] * wk[ky * 9 + kx];
  }
  h[idx] = f2bf(fmaxf(acc, 0.0f));
}

// ---------------------------------------------------------------------------
// PrimaryCaps implicit GEMM:  P[256 x 18432] = W[256 x 5184] * im2col(h)
//   N col n -> (b, oy, ox); K row k -> (ic, ky, kx); stride-2 valid conv.
// Tiles: BM=128 BN=64 BK=32, 8 waves, 2x2 WMMA subtiles per wave.
// A tile (contiguous 128x32 bf16) comes in via the Tensor Data Mover.
__global__ __launch_bounds__(256) void k_pcap_gemm(
    const unsigned short* __restrict__ Aw,   // [256][5184] bf16 weights
    const unsigned short* __restrict__ Hb,   // [32][64][56][56] bf16
    const float* __restrict__ pb,            // [256] bias
    float* __restrict__ P)                   // [32][256][576] fp32
{
  __shared__ __align__(16) unsigned short As[128 * 32];  // [m][k]
  __shared__ __align__(16) unsigned short Bs[64 * 32];   // [n][k]
  const int t     = threadIdx.x;
  const int lane  = t & 31;
  const int wv    = t >> 5;
  const int waveM = wv & 3;
  const int waveN = wv >> 2;
  const int Nbase = blockIdx.x * 64;
  const int Mbase = blockIdx.y * 128;

  // B gather ownership: kk = t&31, 8 consecutive n
  const int kk   = t & 31;
  const int nblk = t >> 5;
  int hbase[8];
  #pragma unroll
  for (int e = 0; e < 8; ++e) {
    int ng = Nbase + nblk * 8 + e;
    int b  = ng / 576;
    int s  = ng - b * 576;
    int oy = s / 24, ox = s - (s / 24) * 24;
    hbase[e] = b * 200704 + (oy * 2) * 56 + ox * 2;
  }

#if USE_TDM
  // LDS byte address of As for the TDM descriptor (addrspace(3) offset).
  const unsigned lds_base =
      (unsigned)(unsigned long long)(__attribute__((address_space(3))) unsigned short*)As;
#else
  const int ar = t >> 1;   // A load ownership: row, 16-element half
  const int ah = t & 1;
#endif

  v8f acc00 = {0,0,0,0,0,0,0,0};
  v8f acc01 = acc00, acc10 = acc00, acc11 = acc00;

  const uint4* Aq = (const uint4*)As;
  const uint4* Bq = (const uint4*)Bs;
  const int kb    = (lane < 16) ? 0 : 8;    // A k-base per lane half
  const int kb2   = (lane < 16) ? 0 : 8;    // B k-base per lane half
  const int col   = lane & 15;
  const int mrow0 = (waveM * 32 + col) * 32;
  const int mrow1 = mrow0 + 16 * 32;
  const int ncol0 = (waveN * 32 + col) * 32;
  const int ncol1 = ncol0 + 16 * 32;

  for (int k0 = 0; k0 < 5184; k0 += 32) {
#if USE_TDM
    if (wv == 0) {
      // D# group0: count=1 | lds_addr | global tile byte addr (57b) | type=2
      unsigned long long ga =
          (unsigned long long)Aw + ((unsigned long long)Mbase * 5184ull + (unsigned)k0) * 2ull;
      u32x4 g0 = { 1u, lds_base, (unsigned)ga, (unsigned)(ga >> 32) | 0x80000000u };
      // D# group1: data_size=2B; tensor 5184x256, tile 32x128, stride0=5184
      i32x8 g1 = { (int)0x00010000,           // wg_mask=0, data_size=1 (2B)
                   (int)0x14400000,           // tensor_dim0[15:0]=5184 << 16
                   (int)0x01000000,           // tensor_dim0 hi=0 | tensor_dim1=256 << 16
                   (int)0x00200000,           // tensor_dim1 hi=0 | tile_dim0=32 << 16
                   (int)128,                  // tile_dim1=128 | tile_dim2=0
                   (int)0x1440,               // tensor_dim0_stride=5184 (lo32)
                   0, 0 };                    // stride0 hi, stride1 (unused 2D)
      i32x4 z4 = {0, 0, 0, 0};
#if defined(__clang_major__) && (__clang_major__ >= 23)
      i32x8 z8 = {0, 0, 0, 0, 0, 0, 0, 0};
      __builtin_amdgcn_tensor_load_to_lds(g0, g1, z4, z4, z8, 0);
#else
      __builtin_amdgcn_tensor_load_to_lds(g0, g1, z4, z4, 0);
#endif
    }
#else
    { // fallback A tile: 128x32 bf16, vectorized 16B copies
      const uint4* src = (const uint4*)(Aw + (size_t)(Mbase + ar) * 5184 + k0 + ah * 16);
      uint4* dst = (uint4*)(As + ar * 32 + ah * 16);
      dst[0] = src[0];
      dst[1] = src[1];
    }
#endif
    { // B tile: im2col gather from h
      int k  = k0 + kk;
      int ic = k / 81;
      int r  = k - ic * 81;
      int ky = r / 9;
      int kx = r - ky * 9;
      int off = ic * 3136 + ky * 56 + kx;
      #pragma unroll
      for (int e = 0; e < 8; ++e)
        Bs[(nblk * 8 + e) * 32 + kk] = Hb[hbase[e] + off];
    }
#if USE_TDM
#if __has_builtin(__builtin_amdgcn_s_wait_tensorcnt)
    __builtin_amdgcn_s_wait_tensorcnt(0);
#else
    asm volatile("s_wait_tensorcnt 0x0" ::: "memory");
#endif
#endif
    __syncthreads();
    Frag16 a0, a1, b0, b1;
    a0.q[0] = Aq[(mrow0 + kb) >> 3];       a0.q[1] = Aq[(mrow0 + 16 + kb) >> 3];
    a1.q[0] = Aq[(mrow1 + kb) >> 3];       a1.q[1] = Aq[(mrow1 + 16 + kb) >> 3];
    b0.q[0] = Bq[(ncol0 + kb2) >> 3];      b0.q[1] = Bq[(ncol0 + kb2 + 16) >> 3];
    b1.q[0] = Bq[(ncol1 + kb2) >> 3];      b1.q[1] = Bq[(ncol1 + kb2 + 16) >> 3];
    acc00 = __builtin_amdgcn_wmma_f32_16x16x32_bf16(false, a0.v, false, b0.v, (short)0, acc00, false, false);
    acc01 = __builtin_amdgcn_wmma_f32_16x16x32_bf16(false, a0.v, false, b1.v, (short)0, acc01, false, false);
    acc10 = __builtin_amdgcn_wmma_f32_16x16x32_bf16(false, a1.v, false, b0.v, (short)0, acc10, false, false);
    acc11 = __builtin_amdgcn_wmma_f32_16x16x32_bf16(false, a1.v, false, b1.v, (short)0, acc11, false, false);
    __syncthreads();
  }

  const int rowadd = (lane < 16) ? 0 : 8;
  #pragma unroll
  for (int tile = 0; tile < 4; ++tile) {
    const v8f& a = (tile == 0) ? acc00 : (tile == 1) ? acc01 : (tile == 2) ? acc10 : acc11;
    int msub = (tile >> 1) * 16;
    int nsub = (tile & 1) * 16;
    int n = Nbase + waveN * 32 + nsub + col;
    int b = n / 576;
    int s = n - b * 576;
    #pragma unroll
    for (int r = 0; r < 8; ++r) {
      int ch = Mbase + waveM * 32 + msub + r + rowadd;
      P[(b * 256 + ch) * 576 + s] = a[r] + pb[ch];
    }
  }
}

// ---------------------------------------------------------------------------
// squash over capsule dim: P[32][256][576] -> U bf16 [v][b][8]  (v = cc*576+s)
__global__ void k_squash_u(const float* __restrict__ P,
                           unsigned short* __restrict__ U) {
  int idx = blockIdx.x * 256 + threadIdx.x;
  if (idx >= 32 * 18432) return;
  int b = idx / 18432;
  int v = idx - b * 18432;
  int cc = v / 576;
  int s  = v - cc * 576;
  const float* p = P + (b * 256 + cc) * 576 + s;
  float pe[8]; float sn = 0.f;
  #pragma unroll
  for (int i = 0; i < 8; ++i) { pe[i] = p[i * 18432]; sn += pe[i] * pe[i]; }
  float scale = __fsqrt_rn(sn) / (1.0f + sn);
  unsigned short* o = U + ((size_t)v * 32 + b) * 8;
  #pragma unroll
  for (int i = 0; i < 8; ++i) o[i] = f2bf(scale * pe[i]);
}

// ---------------------------------------------------------------------------
// pred[v][c][b][o] = sum_i u[v][b][i] * route_w[c][v][i][o], bf16 out.
// One block per v; 20 WMMAs (2 b-tiles x 10 classes), K=8 zero-padded to 32.
__global__ __launch_bounds__(256) void k_pred(
    const unsigned short* __restrict__ U,    // [18432][32][8]
    const unsigned short* __restrict__ Rw,   // [10][18432][8][16]
    unsigned short* __restrict__ Pred)       // [18432][10][32][16]
{
  __shared__ __align__(16) unsigned short Ul[256];
  __shared__ __align__(16) unsigned short Rl[1280];
  const int v = blockIdx.x;
  const int t = threadIdx.x;
  Ul[t] = U[(size_t)v * 256 + t];
  for (int i = t; i < 1280; i += 256) {
    int c = i >> 7;
    int rest = i & 127;
    Rl[i] = Rw[((size_t)c * 18432 + v) * 128 + rest];
  }
  __syncthreads();
  const int lane = t & 31;
  const int wv   = t >> 5;
  const int col  = lane & 15;
  const int kbA  = (lane < 16) ? 0 : 8;
  const int kbB  = (lane < 16) ? 0 : 16;
  const int rowadd = (lane < 16) ? 0 : 8;
  for (int p = wv; p < 20; p += 8) {
    int mtile = p & 1;
    int c = p >> 1;
    Frag16 a, bb;
    #pragma unroll
    for (int j = 0; j < 16; ++j) {
      int Ka = (j < 8) ? (kbA + j) : (16 + kbA + (j - 8));
      a.s[j] = (Ka < 8) ? Ul[(mtile * 16 + col) * 8 + Ka] : (unsigned short)0;
      int Kb = Ka;
      bb.s[j] = (Kb < 8) ? Rl[c * 128 + Kb * 16 + col] : (unsigned short)0;
    }
    v8f acc = {0,0,0,0,0,0,0,0};
    acc = __builtin_amdgcn_wmma_f32_16x16x32_bf16(false, a.v, false, bb.v, (short)0, acc, false, false);
    #pragma unroll
    for (int r = 0; r < 8; ++r) {
      int b = mtile * 16 + r + rowadd;
      Pred[(((size_t)v * 10 + c) * 32 + b) * 16 + col] = f2bf(acc[r]);
    }
  }
}

// ---------------------------------------------------------------------------
// routing: softmax over V stats per (c,b) column
__global__ __launch_bounds__(256) void k_route_stats(const float* __restrict__ Blog,
                                                     float* __restrict__ cbmax,
                                                     float* __restrict__ cbsum) {
  __shared__ float red[256];
  const int col = blockIdx.x;   // c*32 + b
  const int t = threadIdx.x;
  float m = -3.4e38f;
  for (int v = t; v < 18432; v += 256) m = fmaxf(m, Blog[(size_t)v * 320 + col]);
  red[t] = m; __syncthreads();
  for (int s = 128; s > 0; s >>= 1) { if (t < s) red[t] = fmaxf(red[t], red[t + s]); __syncthreads(); }
  const float mx = red[0]; __syncthreads();
  float acc = 0.f;
  for (int v = t; v < 18432; v += 256) acc += __expf(Blog[(size_t)v * 320 + col] - mx);
  red[t] = acc; __syncthreads();
  for (int s = 128; s > 0; s >>= 1) { if (t < s) red[t] += red[t + s]; __syncthreads(); }
  if (t == 0) { cbmax[col] = mx; cbsum[col] = red[0]; }
}

// Sraw[c][b][o] = sum_v pred * softmax_v(Blog)
__global__ __launch_bounds__(256) void k_route_S(const unsigned short* __restrict__ Pred,
                                                 const float* __restrict__ Blog,
                                                 const float* __restrict__ cbmax,
                                                 const float* __restrict__ cbsum,
                                                 float* __restrict__ Sraw) {
  __shared__ float red[256];
  const int col = blockIdx.x;
  const int t = threadIdx.x;
  const int o = t & 15;
  const int g = t >> 4;
  const float mx  = cbmax[col];
  const float inv = 1.0f / cbsum[col];
  float acc = 0.f;
  for (int v = g; v < 18432; v += 16) {
    float wgt = __expf(Blog[(size_t)v * 320 + col] - mx);
    acc += wgt * bf2f(Pred[((size_t)v * 320 + col) * 16 + o]);
  }
  red[t] = acc; __syncthreads();
  for (int s = 128; s >= 16; s >>= 1) { if (t < s) red[t] += red[t + s]; __syncthreads(); }
  if (t < 16) Sraw[col * 16 + t] = red[t] * inv;
}

__global__ void k_squash_S(const float* __restrict__ Sraw, float* __restrict__ S) {
  int col = threadIdx.x;
  if (col >= 320) return;
  float e[16]; float sn = 0.f;
  #pragma unroll
  for (int o = 0; o < 16; ++o) { e[o] = Sraw[col * 16 + o]; sn += e[o] * e[o]; }
  float scale = __fsqrt_rn(sn) / (1.0f + sn);
  #pragma unroll
  for (int o = 0; o < 16; ++o) S[col * 16 + o] = scale * e[o];
}

__global__ void k_blog_update(const unsigned short* __restrict__ Pred,
                              const float* __restrict__ S,
                              float* __restrict__ Blog) {
  size_t idx = (size_t)blockIdx.x * 256 + threadIdx.x;
  if (idx >= (size_t)18432 * 320) return;
  int col = (int)(idx % 320);
  float d = 0.f;
  #pragma unroll
  for (int o = 0; o < 16; ++o) d += bf2f(Pred[idx * 16 + o]) * S[col * 16 + o];
  Blog[idx] += d;
}

// ---------------------------------------------------------------------------
// classes softmax + argmax mask -> decoder input
__global__ void k_finalize(const float* __restrict__ S, float* __restrict__ classes,
                           float* __restrict__ dec_in) {
  int b = threadIdx.x;
  if (b >= 32) return;
  float len[10];
  float mx = -3.4e38f;
  for (int c = 0; c < 10; ++c) {
    float sn = 0.f;
    for (int o = 0; o < 16; ++o) { float x = S[(c * 32 + b) * 16 + o]; sn += x * x; }
    len[c] = __fsqrt_rn(sn);
    mx = fmaxf(mx, len[c]);
  }
  float sum = 0.f;
  for (int c = 0; c < 10; ++c) { len[c] = __expf(len[c] - mx); sum += len[c]; }
  int amax = 0; float best = -1.f;
  for (int c = 0; c < 10; ++c) {
    float cl = len[c] / sum;
    classes[b * 10 + c] = cl;
    if (cl > best) { best = cl; amax = c; }
  }
  for (int c = 0; c < 10; ++c)
    for (int o = 0; o < 16; ++o)
      dec_in[b * 160 + c * 16 + o] = (c == amax) ? S[(c * 32 + b) * 16 + o] : 0.f;
}

// dense: out[b][j] = act(in[b]·W[:,j] + bias[j]); act 0=relu 1=sigmoid
__global__ void k_dense(const float* __restrict__ in, const float* __restrict__ W,
                        const float* __restrict__ bias, float* __restrict__ out,
                        int K, int N, int act) {
  int idx = blockIdx.x * 256 + threadIdx.x;
  if (idx >= 32 * N) return;
  int b = idx / N;
  int j = idx - b * N;
  float acc = bias[j];
  for (int k = 0; k < K; ++k) acc += in[b * K + k] * W[(size_t)k * N + j];
  if (act == 0) acc = fmaxf(acc, 0.f);
  else          acc = 1.0f / (1.0f + __expf(-acc));
  out[idx] = acc;
}

// ---------------------------------------------------------------------------
extern "C" void kernel_launch(void* const* d_in, const int* in_sizes, int n_in,
                              void* d_out, int out_size, void* d_ws, size_t ws_size,
                              hipStream_t stream) {
  (void)in_sizes; (void)n_in; (void)out_size; (void)ws_size;
  const float* x       = (const float*)d_in[0];
  const float* conv1_w = (const float*)d_in[1];
  const float* conv1_b = (const float*)d_in[2];
  const float* pcap_w  = (const float*)d_in[3];
  const float* pcap_b  = (const float*)d_in[4];
  const float* route_w = (const float*)d_in[5];
  const float* dec_w1  = (const float*)d_in[6];
  const float* dec_b1  = (const float*)d_in[7];
  const float* dec_w2  = (const float*)d_in[8];
  const float* dec_b2  = (const float*)d_in[9];
  const float* dec_w3  = (const float*)d_in[10];
  const float* dec_b3  = (const float*)d_in[11];
  float* out = (float*)d_out;

  char* ws = (char*)d_ws;
  size_t off = 0;
  auto alloc = [&](size_t bytes) -> void* {
    void* p = ws + off;
    off = (off + bytes + 255) & ~(size_t)255;
    return p;
  };
  unsigned short* hbf    = (unsigned short*)alloc((size_t)6422528 * 2);   // conv1 out bf16
  unsigned short* pcapwb = (unsigned short*)alloc((size_t)1327104 * 2);   // pcap W bf16
  float*          P      = (float*)         alloc((size_t)4718592 * 4);   // pcap conv out
  unsigned short* U      = (unsigned short*)alloc((size_t)4718592 * 2);   // squashed u bf16
  unsigned short* Rwb    = (unsigned short*)alloc((size_t)23592960 * 2);  // route W bf16
  unsigned short* Pred   = (unsigned short*)alloc((size_t)94371840 * 2);  // pred bf16
  float*          Blog   = (float*)         alloc((size_t)5898240 * 4);   // routing logits
  float*          cbmax  = (float*)         alloc(320 * 4);
  float*          cbsum  = (float*)         alloc(320 * 4);
  float*          Sraw   = (float*)         alloc(5120 * 4);
  float*          S      = (float*)         alloc(5120 * 4);
  float*          decin  = (float*)         alloc(5120 * 4);
  float*          h1     = (float*)         alloc(16384 * 4);
  float*          h2     = (float*)         alloc(32768 * 4);

  k_f32_to_bf16<<<5184,  256, 0, stream>>>(pcap_w,  pcapwb, 1327104);
  k_f32_to_bf16<<<92160, 256, 0, stream>>>(route_w, Rwb,    23592960);
  k_conv1_relu <<<25088, 256, 0, stream>>>(x, conv1_w, conv1_b, hbf);
  k_pcap_gemm  <<<dim3(288, 2), 256, 0, stream>>>(pcapwb, hbf, pcap_b, P);
  k_squash_u   <<<2304,  256, 0, stream>>>(P, U);
  k_pred       <<<18432, 256, 0, stream>>>(U, Rwb, Pred);
  k_zero       <<<23040, 256, 0, stream>>>(Blog, 5898240);

  for (int it = 0; it < 3; ++it) {
    k_route_stats<<<320, 256, 0, stream>>>(Blog, cbmax, cbsum);
    k_route_S    <<<320, 256, 0, stream>>>(Pred, Blog, cbmax, cbsum, Sraw);
    k_squash_S   <<<1,   320, 0, stream>>>(Sraw, S);
    if (it < 2)
      k_blog_update<<<23040, 256, 0, stream>>>(Pred, S, Blog);
  }

  k_finalize<<<1, 32, 0, stream>>>(S, out, decin);
  k_dense<<<64,  256, 0, stream>>>(decin, dec_w1, dec_b1, h1,        160,  512,  0);
  k_dense<<<128, 256, 0, stream>>>(h1,    dec_w2, dec_b2, h2,        512,  1024, 0);
  k_dense<<<512, 256, 0, stream>>>(h2,    dec_w3, dec_b3, out + 320, 1024, 4096, 1);
}
